// molecular_network_56410100466094
// MI455X (gfx1250) — matrix operand order repaired and checked by reading, not compiled, not run on hardware
//
#include <hip/hip_runtime.h>
#include <hip/hip_bf16.h>

#define N_NODES 50000
#define N_EDGES 400000

#define INV_M     0.17677669529663687f   // 1/sqrt(32)
#define INV_2M    0.125f                 // 1/sqrt(64)
#define INV_NN    0.35355339059327373f   // 1/sqrt(8)
#define INV_NB    0.22360679774997896f   // 1/sqrt(20)
#define INV_HID   0.1f                   // 1/sqrt(100)
#define INV_SQRT3 0.57735026918962576f

typedef __attribute__((ext_vector_type(16))) _Float16 v16h;
typedef __attribute__((ext_vector_type(8)))  float    v8f;

union V16 { v16h v; unsigned u[8]; };
struct __align__(16) U4 { unsigned x, y, z, w; };

__device__ __forceinline__ unsigned pkh2(float a, float b) {
  union { _Float16 h[2]; unsigned u; } t;
  t.h[0] = (_Float16)a; t.h[1] = (_Float16)b;
  return t.u;
}

// ---------------------------------------------------------------------------
// Kernel 0: zero the aggregation buffer
// ---------------------------------------------------------------------------
__global__ void zero_kernel(float* __restrict__ p, long n) {
  long i = (long)blockIdx.x * blockDim.x + threadIdx.x;
  long stride = (long)gridDim.x * blockDim.x;
  for (; i < n; i += stride) p[i] = 0.f;
}

// ---------------------------------------------------------------------------
// Kernel 1: per-node s0 = x0@W_si0 * inv_m * a ; s1c = x1[:,:,c]@W_si1 * inv_m * a
// snode layout: [node][0..31]=s0, [node][32+c*32+m]=s1[m][c]
// ---------------------------------------------------------------------------
__global__ __launch_bounds__(256) void node_prep(
    const float* __restrict__ ninp, const float* __restrict__ nattr,
    const float* __restrict__ Wsi0, const float* __restrict__ Wsi1,
    float* __restrict__ snode)
{
  __shared__ float ws0[1024], ws1[1024];
  __shared__ __align__(16) float xs[32][128];
  const int tid = threadIdx.x;
  for (int i = tid; i < 1024; i += 256) { ws0[i] = Wsi0[i]; ws1[i] = Wsi1[i]; }
  const int wave = tid >> 5, lane = tid & 31;

  for (int chunk = 0; chunk < 8; ++chunk) {
    const int nbase = blockIdx.x * 256 + chunk * 32;
    __syncthreads();
    for (int i = tid; i < 1024; i += 256) {     // 32 rows x 32 float4
      int r = i >> 5, q = i & 31;
      int node = nbase + r;
      float4 v = make_float4(0.f, 0.f, 0.f, 0.f);
      if (node < N_NODES) v = *(const float4*)(ninp + (size_t)node * 128 + q * 4);
      *(float4*)&xs[r][q * 4] = v;
    }
    __syncthreads();
    #pragma unroll
    for (int t = 0; t < 4; ++t) {
      int r = wave * 4 + t;
      int node = nbase + r;
      if (node >= N_NODES) continue;
      const float* x = xs[r];
      float s0 = 0.f, s10 = 0.f, s11 = 0.f, s12 = 0.f;
      #pragma unroll
      for (int k = 0; k < 32; ++k) {
        float w0 = ws0[k * 32 + lane];
        float w1 = ws1[k * 32 + lane];
        s0  += x[k]          * w0;
        s10 += x[32 + 3 * k] * w1;
        s11 += x[33 + 3 * k] * w1;
        s12 += x[34 + 3 * k] * w1;
      }
      float sc = nattr[node] * INV_M;
      float* sp = snode + (size_t)node * 128;
      sp[lane]      = s0  * sc;
      sp[32 + lane] = s10 * sc;
      sp[64 + lane] = s11 * sc;
      sp[96 + lane] = s12 * sc;
    }
  }
}

// ---------------------------------------------------------------------------
// Kernel 2: fused edge pipeline.
//   w = (silu(de @ W1 * inv_nb) @ W2) * inv_hid   via v_wmma_f32_16x16x32_f16
//   then gather s[dst], form messages, atomic-scatter into agg[src].
// agg layout: [node][0..31]=mid0a, [32..63]=mid0b, [64+c*64+m]=m1a, [64+c*64+32+m]=m1b
// ---------------------------------------------------------------------------
__global__ __launch_bounds__(256) void edge_kernel(
    const float* __restrict__ dist, const float* __restrict__ W1,
    const float* __restrict__ W2,   const int* __restrict__ esrc,
    const int* __restrict__ edst,   const float* __restrict__ eattr,
    const float* __restrict__ snode, float* __restrict__ agg)
{
  __shared__ __align__(16) unsigned w1frag[7][32][8];       // B frags, GEMM1 (K=32 pad, 7 N tiles)
  __shared__ __align__(16) unsigned w2frag[4][8][32][8];    // B frags, GEMM2 (4 K tiles, 8 N tiles)
  __shared__ __align__(16) float    wsbuf[8][16][128];      // per-wave w staging; aliased as f16 h staging

  const int tid = threadIdx.x;

  // Pack W_mlp1 (20x100 -> 32x112 zero-padded) into 16-bit B-fragment layout:
  // lane<16: col=lane, K=0..7 in j 0..3, K=16..23 in j 4..7 ; lane>=16: col=lane-16, K=8..15, 24..31 (low half = even K)
  for (int idx = tid; idx < 7 * 32 * 8; idx += 256) {
    int j = idx & 7, ln = (idx >> 3) & 31, nt = idx >> 8;
    int n = nt * 16 + (ln & 15);
    int k = ((j >> 2) << 4) + ((ln >> 4) << 3) + 2 * (j & 3);
    float v0 = (k     < 20 && n < 100) ? W1[k * 100 + n]       : 0.f;
    float v1 = (k + 1 < 20 && n < 100) ? W1[(k + 1) * 100 + n] : 0.f;
    w1frag[nt][ln][j] = pkh2(v0, v1);
  }
  // Pack W_mlp2 (100x128 -> 128x128 zero-padded along K)
  for (int idx = tid; idx < 4 * 8 * 32 * 8; idx += 256) {
    int j = idx & 7, ln = (idx >> 3) & 31, nt = (idx >> 8) & 7, kt = idx >> 11;
    int n = nt * 16 + (ln & 15);
    int k = kt * 32 + ((j >> 2) << 4) + ((ln >> 4) << 3) + 2 * (j & 3);
    float v0 = (k     < 100) ? W2[k * 128 + n]       : 0.f;
    float v1 = (k + 1 < 100) ? W2[(k + 1) * 128 + n] : 0.f;
    w2frag[kt][nt][ln][j] = pkh2(v0, v1);
  }
  __syncthreads();

  const int wave = tid >> 5, lane = tid & 31;
  const int l15 = lane & 15, half = lane >> 4;
  const long ebase = (long)blockIdx.x * 128 + wave * 16;

  _Float16* hb = (_Float16*)&wsbuf[wave][0][0];   // 16x128 f16 region, wave-private

  // --- A fragment for GEMM1: 16 edges x K=20 (padded to 32)
  // lane<16: K=0..7 in u[0..3], K=16..23 in u[4..7]; lane>=16: K=8..15, K=24..31
  const float* dep = dist + (ebase + l15) * 20;   // rows are 80B => 16B aligned
  float4 p0 = *(const float4*)(dep + half * 8);
  float4 p1 = *(const float4*)(dep + half * 8 + 4);
  float4 p2 = *(const float4*)(dep + 16);
  V16 a1;
  a1.u[0] = pkh2(p0.x, p0.y);  a1.u[1] = pkh2(p0.z, p0.w);
  a1.u[2] = pkh2(p1.x, p1.y);  a1.u[3] = pkh2(p1.z, p1.w);
  a1.u[4] = half ? 0u : pkh2(p2.x, p2.y);
  a1.u[5] = half ? 0u : pkh2(p2.z, p2.w);
  a1.u[6] = 0u;  a1.u[7] = 0u;

  // --- GEMM1 + silu -> hb (f16), cols 0..111 (+ zero pad 112..127)
  for (int nt = 0; nt < 7; ++nt) {
    V16 b;
    const U4* bp = (const U4*)&w1frag[nt][lane][0];
    U4 q0 = bp[0], q1 = bp[1];
    b.u[0] = q0.x; b.u[1] = q0.y; b.u[2] = q0.z; b.u[3] = q0.w;
    b.u[4] = q1.x; b.u[5] = q1.y; b.u[6] = q1.z; b.u[7] = q1.w;
    v8f c = {};
    c = __builtin_amdgcn_wmma_f32_16x16x32_f16(false, a1.v, false, b.v, (short)0, c, false, false);
    #pragma unroll
    for (int r = 0; r < 8; ++r) {               // row = r + 8*half, col = nt*16 + l15
      float x = c[r] * INV_NB;
      // silu via fast reciprocal (single v_rcp_f32 instead of IEEE div sequence)
      float s = x * __fdividef(1.f, 1.f + __expf(-x));
      hb[(r + 8 * half) * 128 + nt * 16 + l15] = (_Float16)s;
    }
  }
  #pragma unroll
  for (int r = 0; r < 8; ++r)                   // zero K-pad cols 112..127
    hb[(r + 8 * half) * 128 + 112 + l15] = (_Float16)0.f;

  // --- Re-layout h into A fragments for GEMM2 (K = 128)
  V16 a2[4];
  #pragma unroll
  for (int kt = 0; kt < 4; ++kt) {
    int c0 = kt * 32 + half * 8;
    U4 q0 = *(const U4*)&hb[l15 * 128 + c0];
    U4 q1 = *(const U4*)&hb[l15 * 128 + c0 + 16];
    a2[kt].u[0] = q0.x; a2[kt].u[1] = q0.y; a2[kt].u[2] = q0.z; a2[kt].u[3] = q0.w;
    a2[kt].u[4] = q1.x; a2[kt].u[5] = q1.y; a2[kt].u[6] = q1.z; a2[kt].u[7] = q1.w;
  }

  // --- GEMM2: 8 N-tiles x 4 chained K-tiles, write w (f32) to wsbuf
  for (int nt = 0; nt < 8; ++nt) {
    v8f c = {};
    #pragma unroll
    for (int kt = 0; kt < 4; ++kt) {
      V16 b;
      const U4* bp = (const U4*)&w2frag[kt][nt][lane][0];
      U4 q0 = bp[0], q1 = bp[1];
      b.u[0] = q0.x; b.u[1] = q0.y; b.u[2] = q0.z; b.u[3] = q0.w;
      b.u[4] = q1.x; b.u[5] = q1.y; b.u[6] = q1.z; b.u[7] = q1.w;
      c = __builtin_amdgcn_wmma_f32_16x16x32_f16(false, a2[kt].v, false, b.v, (short)0, c, false, false);
    }
    #pragma unroll
    for (int r = 0; r < 8; ++r)
      wsbuf[wave][r + 8 * half][nt * 16 + l15] = c[r] * INV_HID;
  }

  // --- Messages + atomic scatter (lane = feature m)
  for (int e = 0; e < 16; ++e) {
    long ge = ebase + e;
    int src = esrc[ge], dst = edst[ge];
    float4 ya = *(const float4*)(eattr + ge * 4);     // y0 = ya.x, y1 = (ya.y, ya.z, ya.w)
    float w1v = wsbuf[wave][e][lane];
    float w2v = wsbuf[wave][e][32 + lane];
    float w3v = wsbuf[wave][e][64 + lane];
    float w4v = wsbuf[wave][e][96 + lane];
    const float* sn = snode + (size_t)dst * 128;
    float g0  = sn[lane];
    float g1x = sn[32 + lane], g1y = sn[64 + lane], g1z = sn[96 + lane];
    float* ap = agg + (size_t)src * 256;
    atomicAdd(ap + lane,        g0 * ya.x * w1v);
    atomicAdd(ap + 32  + lane,  (g1x * ya.y + g1y * ya.z + g1z * ya.w) * INV_SQRT3 * w4v);
    atomicAdd(ap + 64  + lane,  g0 * ya.y * w2v);
    atomicAdd(ap + 96  + lane,  g1x * ya.x * w3v);
    atomicAdd(ap + 128 + lane,  g0 * ya.z * w2v);
    atomicAdd(ap + 160 + lane,  g1y * ya.x * w3v);
    atomicAdd(ap + 192 + lane,  g0 * ya.w * w2v);
    atomicAdd(ap + 224 + lane,  g1z * ya.x * w3v);
  }
}

// ---------------------------------------------------------------------------
// Kernel 3: per-node output GEMMs + skip connection
// ---------------------------------------------------------------------------
__global__ __launch_bounds__(256) void node_out(
    const float* __restrict__ ninp, const float* __restrict__ nattr,
    const float* __restrict__ Wout0, const float* __restrict__ Wout1,
    const float* __restrict__ Wsc0,  const float* __restrict__ Wsc1,
    const float* __restrict__ agg,   float* __restrict__ out)
{
  __shared__ float wo0[2048], wo1[2048], wc0[1024], wc1[1024];
  __shared__ __align__(16) float as_[16][256];
  __shared__ __align__(16) float xs[16][128];
  const int tid = threadIdx.x;
  for (int i = tid; i < 2048; i += 256) { wo0[i] = Wout0[i]; wo1[i] = Wout1[i]; }
  for (int i = tid; i < 1024; i += 256) { wc0[i] = Wsc0[i];  wc1[i] = Wsc1[i]; }
  const int wave = tid >> 5, lane = tid & 31;

  for (int chunk = 0; chunk < 8; ++chunk) {
    const int nbase = blockIdx.x * 128 + chunk * 16;
    __syncthreads();
    for (int i = tid; i < 1024; i += 256) {     // agg rows: 16 x 64 float4
      int r = i >> 6, q = i & 63;
      int node = nbase + r;
      float4 v = make_float4(0.f, 0.f, 0.f, 0.f);
      if (node < N_NODES) v = *(const float4*)(agg + (size_t)node * 256 + q * 4);
      *(float4*)&as_[r][q * 4] = v;
    }
    for (int i = tid; i < 512; i += 256) {      // x rows: 16 x 32 float4
      int r = i >> 5, q = i & 31;
      int node = nbase + r;
      float4 v = make_float4(0.f, 0.f, 0.f, 0.f);
      if (node < N_NODES) v = *(const float4*)(ninp + (size_t)node * 128 + q * 4);
      *(float4*)&xs[r][q * 4] = v;
    }
    __syncthreads();
    #pragma unroll
    for (int t = 0; t < 2; ++t) {
      int r = wave * 2 + t;
      int node = nbase + r;
      if (node >= N_NODES) continue;
      float acc0 = 0.f, b0 = 0.f, b1 = 0.f, b2 = 0.f;
      #pragma unroll
      for (int k = 0; k < 64; ++k) {
        float u0 = wo0[k * 32 + lane], u1 = wo1[k * 32 + lane];
        acc0 += as_[r][k]       * u0;
        b0   += as_[r][64  + k] * u1;
        b1   += as_[r][128 + k] * u1;
        b2   += as_[r][192 + k] * u1;
      }
      float sc0 = 0.f, t0 = 0.f, t1 = 0.f, t2 = 0.f;
      #pragma unroll
      for (int k = 0; k < 32; ++k) {
        float c0 = wc0[k * 32 + lane], c1 = wc1[k * 32 + lane];
        sc0 += xs[r][k]          * c0;
        t0  += xs[r][32 + 3 * k] * c1;
        t1  += xs[r][33 + 3 * k] * c1;
        t2  += xs[r][34 + 3 * k] * c1;
      }
      float a_n = nattr[node];
      float fo = INV_NN * INV_2M * a_n;
      float fs = INV_M * a_n;
      float* op = out + (size_t)node * 128;
      op[lane]              = acc0 * fo + sc0 * fs;
      op[32 + lane * 3 + 0] = b0 * fo + t0 * fs;
      op[32 + lane * 3 + 1] = b1 * fo + t1 * fs;
      op[32 + lane * 3 + 2] = b2 * fo + t2 * fs;
    }
  }
}

// ---------------------------------------------------------------------------
extern "C" void kernel_launch(void* const* d_in, const int* in_sizes, int n_in,
                              void* d_out, int out_size, void* d_ws, size_t ws_size,
                              hipStream_t stream) {
  (void)in_sizes; (void)n_in; (void)out_size; (void)ws_size;
  const float* node_input = (const float*)d_in[0];
  const float* node_attr  = (const float*)d_in[1];
  const int*   edge_src   = (const int*)d_in[2];
  const int*   edge_dst   = (const int*)d_in[3];
  const float* edge_attr  = (const float*)d_in[4];
  const float* dist_emb   = (const float*)d_in[5];
  const float* W_si0  = (const float*)d_in[6];
  const float* W_si1  = (const float*)d_in[7];
  const float* W_mlp1 = (const float*)d_in[8];
  const float* W_mlp2 = (const float*)d_in[9];
  const float* W_out0 = (const float*)d_in[10];
  const float* W_out1 = (const float*)d_in[11];
  const float* W_sc0  = (const float*)d_in[12];
  const float* W_sc1  = (const float*)d_in[13];
  float* out   = (float*)d_out;
  float* snode = (float*)d_ws;                           // N * 128 f32
  float* agg   = snode + (size_t)N_NODES * 128;          // N * 256 f32

  zero_kernel<<<2048, 256, 0, stream>>>(agg, (long)N_NODES * 256);
  node_prep<<<(N_NODES + 255) / 256, 256, 0, stream>>>(node_input, node_attr, W_si0, W_si1, snode);
  edge_kernel<<<N_EDGES / 128, 256, 0, stream>>>(dist_emb, W_mlp1, W_mlp2,
                                                 edge_src, edge_dst, edge_attr, snode, agg);
  node_out<<<(N_NODES + 127) / 128, 256, 0, stream>>>(node_input, node_attr,
                                                      W_out0, W_out1, W_sc0, W_sc1, agg, out);
}
